// CNNGRUforHorizon_60687887892615
// MI455X (gfx1250) — matrix operand order repaired and checked by reading, not compiled, not run on hardware
//
#include <hip/hip_runtime.h>
#include <hip/hip_bf16.h>
#include <math.h>

typedef __attribute__((ext_vector_type(16))) _Float16 v16h;
typedef __attribute__((ext_vector_type(8)))  _Float16 v8h;
typedef __attribute__((ext_vector_type(8)))  float    v8f;

union V16H { v16h v; v8h h[2]; };

#define LDSB_STRIDE 40  // halves per B row: 32 data + 8 pad -> 80 B, 16 B aligned, bank-conflict-free

// ---------------------------------------------------------------------------
// WMMA implicit-GEMM conv (NHWC f16 act, [tap][Cout][Cin] f16 weights)
// out[m][n] = epilogue( sum_{t,k} act[(y+dy)(x+dx)][k] * wt[t][n][k] )
// epilogue: v = acc*scale[n] + shift[n]; optional relu; f16 or f32 store.
// Block = 128 threads (4 waves along M). Each wave computes a 32(M) x 64(N)
// tile (2 A fragments x 4 B fragments = 8 WMMA per 32-K step). The 32x64 f16
// B tile is shared by all 4 waves, so it is staged once per block in LDS.
// Requires: Cin % 32 == 0, Cout % 64 == 0, M % 16 == 0, taps in {1,9}.
// ---------------------------------------------------------------------------
__global__ void conv_wmma_kernel(const _Float16* __restrict__ act,
                                 const _Float16* __restrict__ wt,
                                 const float* __restrict__ scale,
                                 const float* __restrict__ shift,
                                 void* __restrict__ out,
                                 int H, int W, int Cin, int Cout,
                                 int taps, int relu, int out_f16)
{
    __shared__ _Float16 ldsB[64 * LDSB_STRIDE];  // 5 KB

    const int tid   = threadIdx.x;
    const int lane  = tid & 31;
    const int wave  = tid >> 5;
    const int lrow  = lane & 15;   // A: row within subtile / B: col within subtile
    const int lhalf = lane >> 4;   // K-group selector
    const int M  = H * W;
    const int m0 = blockIdx.x * 128 + wave * 32;
    const int n0 = blockIdx.y * 64;

    // Spatial coords for the two M-subtiles of this wave (constant all kernel)
    int ay[2], ax[2];
    bool mval[2];
#pragma unroll
    for (int u = 0; u < 2; ++u) {
        const int m = m0 + u * 16 + lrow;
        mval[u] = (m < M);
        const int mc = mval[u] ? m : 0;
        ay[u] = mc / W;
        ax[u] = mc - ay[u] * W;
    }

    // Cooperative B-tile load mapping: thread -> (row in tile, 16-halve chunk)
    const int bn = tid >> 1;          // 0..63
    const int bk = (tid & 1) * 16;    // 0 or 16

    v8f acc[8];
#pragma unroll
    for (int s = 0; s < 8; ++s) { v8f z = {}; acc[s] = z; }

    for (int t = 0; t < taps; ++t) {
        int dy = 0, dx = 0;
        if (taps == 9) { dy = t / 3 - 1; dx = t - (t / 3) * 3 - 1; }
        const _Float16* __restrict__ wrow = wt + (size_t)t * Cout * Cin;

        for (int kb = 0; kb < Cin; kb += 32) {
            // ---- stage B tile (rows n0..n0+63, k = kb..kb+31) in LDS ----
            const _Float16* q = wrow + (size_t)(n0 + bn) * Cin + kb + bk;
            const v8h bg0 = *(const v8h*)q;
            const v8h bg1 = *(const v8h*)(q + 8);
            __syncthreads();   // previous tile fully consumed
            *(v8h*)&ldsB[bn * LDSB_STRIDE + bk]     = bg0;
            *(v8h*)&ldsB[bn * LDSB_STRIDE + bk + 8] = bg1;
            __syncthreads();   // tile ready

            // ---- A fragments (2 M-subtiles), implicit im2col with zero pad ----
            v16h af[2];
#pragma unroll
            for (int u = 0; u < 2; ++u) {
                const int yy = ay[u] + dy, xx = ax[u] + dx;
                const bool inb = mval[u] && yy >= 0 && yy < H && xx >= 0 && xx < W;
                V16H a;
                if (inb) {
                    const _Float16* p = act + (size_t)(yy * W + xx) * Cin + kb + lhalf * 8;
                    a.h[0] = *(const v8h*)p;        // k = kb + lhalf*8 + [0..7]
                    a.h[1] = *(const v8h*)(p + 16); // k = kb + lhalf*8 + 16 + [0..7]
                } else {
                    v16h z = {};
                    a.v = z;
                }
                af[u] = a.v;
            }

            // ---- B fragments from LDS + 8 WMMA ----
#pragma unroll
            for (int s = 0; s < 4; ++s) {
                V16H b;
                const _Float16* bp = &ldsB[(s * 16 + lrow) * LDSB_STRIDE + lhalf * 8];
                b.h[0] = *(const v8h*)bp;
                b.h[1] = *(const v8h*)(bp + 16);
                acc[s]     = __builtin_amdgcn_wmma_f32_16x16x32_f16(
                                 false, af[0], false, b.v, (short)0, acc[s],     false, false);
                acc[4 + s] = __builtin_amdgcn_wmma_f32_16x16x32_f16(
                                 false, af[1], false, b.v, (short)0, acc[4 + s], false, false);
            }
        }
    }

    // Epilogue. D layout: VGPR r -> row m0 + u*16 + r + 8*lhalf, col n0 + s*16 + lrow
    float* outf = (float*)out;
    _Float16* outh = (_Float16*)out;
#pragma unroll
    for (int s = 0; s < 4; ++s) {
        const int n = n0 + s * 16 + lrow;
        const float sc = scale[n];
        const float sh = shift[n];
#pragma unroll
        for (int u = 0; u < 2; ++u) {
            if (!mval[0]) continue;           // whole-wave-uniform subtile guards
            if (u == 1 && m0 + 16 >= M) continue;
#pragma unroll
            for (int r = 0; r < 8; ++r) {
                const int mm = m0 + u * 16 + r + lhalf * 8;
                float v = acc[u * 4 + s][r] * sc + sh;
                if (relu) v = fmaxf(v, 0.0f);
                const size_t oi = (size_t)mm * Cout + n;
                if (out_f16) outh[oi] = (_Float16)v;
                else         outf[oi] = v;
            }
        }
    }
}

// ---------------------------------------------------------------------------
// Weight repack: OIHW f32 -> [ky*KW+kx][O][I] f16
// ---------------------------------------------------------------------------
__global__ void convert_w_kernel(const float* __restrict__ w, _Float16* __restrict__ out,
                                 int O, int I, int KH, int KW)
{
    const int idx = blockIdx.x * blockDim.x + threadIdx.x;
    const int total = O * I * KH * KW;
    if (idx >= total) return;
    const int kx = idx % KW;
    int t1 = idx / KW;
    const int ky = t1 % KH;
    int t2 = t1 / KH;
    const int i = t2 % I;
    const int o = t2 / I;
    out[((size_t)(ky * KW + kx) * O + o) * I + i] = (_Float16)w[idx];
}

// ---------------------------------------------------------------------------
// Per-output-channel epilogue scale/shift tables (folded BN / bias)
// ss layout (floats): sc2[128]@0 sh2[128]@128 sc3[256]@256 sh3[256]@512
//                     scfc[512]@768 shfc[512]@1280 scg[384]@1792
//                     shgf[384]@2176 shgb[384]@2560   (total 2944)
// ---------------------------------------------------------------------------
__global__ void setup_ss_kernel(const float* mb2, const float* g2, const float* b2,
                                const float* m2, const float* v2,
                                const float* mb3, const float* fb,
                                const float* bihf, const float* bihb,
                                float* __restrict__ ss)
{
    const int i = threadIdx.x;
    if (i < 128) {
        const float inv = g2[i] * rsqrtf(v2[i] + 1e-5f);
        ss[i]       = inv;
        ss[128 + i] = mb2[i] * inv + b2[i] - m2[i] * inv;
    }
    if (i < 256) { ss[256 + i] = 1.0f; ss[512 + i] = mb3[i]; }
    if (i < 512) { ss[768 + i] = 1.0f; ss[1280 + i] = fb[i]; }
    if (i < 384) { ss[1792 + i] = 1.0f; ss[2176 + i] = bihf[i]; ss[2560 + i] = bihb[i]; }
}

// ---------------------------------------------------------------------------
// Wave branch: generic 1-D conv (pad/stride) + ReLU + maxpool.
// in: [Cin][Lin] f32, w: (Cout,Cin,1,K) OIHW, out: [Cout][Lconv/pool] f32.
// ---------------------------------------------------------------------------
__global__ void wave_conv_pool_kernel(const float* __restrict__ in,
                                      const float* __restrict__ w,
                                      const float* __restrict__ bias,
                                      float* __restrict__ out,
                                      int Cin, int Lin, int Cout, int Lconv,
                                      int K, int stride, int pad, int pool)
{
    const int Lout = Lconv / pool;
    const int tid = blockIdx.x * blockDim.x + threadIdx.x;
    if (tid >= Cout * Lout) return;
    const int c  = tid / Lout;
    const int j2 = tid - c * Lout;
    float mx = 0.0f; // ReLU output is >= 0, so 0 is a safe identity for max
    for (int p = 0; p < pool; ++p) {
        const int j = j2 * pool + p;
        float s = bias[c];
        for (int ci = 0; ci < Cin; ++ci) {
            const float* inr = in + (size_t)ci * Lin;
            const float* wr  = w + ((size_t)c * Cin + ci) * K;
            for (int i = 0; i < K; ++i) {
                const int pos = j * stride + i - pad;
                if (pos >= 0 && pos < Lin) s += inr[pos] * wr[i];
            }
        }
        mx = fmaxf(mx, fmaxf(s, 0.0f));
    }
    out[(size_t)c * Lout + j2] = mx;
}

// ---------------------------------------------------------------------------
// mel conv1: 1->64, 3x3, pad 1, fused BN+ReLU, writes NHWC f16 [64*1024][64]
// ---------------------------------------------------------------------------
__global__ void mel_conv1_kernel(const float* __restrict__ logmel,
                                 const float* __restrict__ w,
                                 const float* __restrict__ bias,
                                 const float* __restrict__ g, const float* __restrict__ be,
                                 const float* __restrict__ mn, const float* __restrict__ vr,
                                 _Float16* __restrict__ act1)
{
    const int tid = blockIdx.x * blockDim.x + threadIdx.x;
    if (tid >= 64 * 1024 * 64) return;
    const int c   = tid & 63;
    const int pos = tid >> 6;
    const int x   = pos & 1023;
    const int y   = pos >> 10;
    float s = bias[c];
    const float* wc = w + c * 9;
    for (int dy = -1; dy <= 1; ++dy) {
        const int yy = y + dy;
        if (yy < 0 || yy >= 64) continue;
        for (int dx = -1; dx <= 1; ++dx) {
            const int xx = x + dx;
            if (xx < 0 || xx >= 1024) continue;
            s += logmel[yy * 1024 + xx] * wc[(dy + 1) * 3 + (dx + 1)];
        }
    }
    const float inv = g[c] * rsqrtf(vr[c] + 1e-5f);
    const float v = s * inv + (be[c] - mn[c] * inv);
    act1[tid] = (_Float16)fmaxf(v, 0.0f);
}

// ---------------------------------------------------------------------------
// Build fused NHWC f16 input for fc: ch 0..255 = bilinear-resized wave branch
// (H:1->32 collapses to replication, W:50->512 half-pixel linear),
// ch 256..511 = 2x2 maxpool of mc3 output (NHWC f16 [64*1024][256]).
// out: [32*512][512] f16 (index == tid).
// ---------------------------------------------------------------------------
__global__ void build_concat_kernel(const float* __restrict__ c4,
                                    const _Float16* __restrict__ act3,
                                    _Float16* __restrict__ cat)
{
    const int tid = blockIdx.x * blockDim.x + threadIdx.x;
    if (tid >= 32 * 512 * 512) return;
    const int c = tid & 511;
    const int x = (tid >> 9) & 511;
    float v;
    if (c < 256) {
        const float s = (x + 0.5f) * (50.0f / 512.0f) - 0.5f;
        const int x0 = (int)floorf(s);
        const float f = s - (float)x0;
        int xa = x0;     xa = xa < 0 ? 0 : (xa > 49 ? 49 : xa);
        int xb = x0 + 1; xb = xb < 0 ? 0 : (xb > 49 ? 49 : xb);
        v = c4[c * 50 + xa] * (1.0f - f) + c4[c * 50 + xb] * f;
    } else {
        const int y  = tid >> 18;
        const int cm = c - 256;
        const _Float16* b = act3 + cm;
        const size_t i00 = (size_t)((2 * y) * 1024 + 2 * x) * 256;
        const size_t i01 = i00 + 256;
        const size_t i10 = i00 + (size_t)1024 * 256;
        const size_t i11 = i10 + 256;
        const float a0 = (float)b[i00], a1 = (float)b[i01];
        const float a2 = (float)b[i10], a3 = (float)b[i11];
        v = fmaxf(fmaxf(a0, a1), fmaxf(a2, a3));
    }
    cat[tid] = (_Float16)v;
}

// ---------------------------------------------------------------------------
// feat[y][c] = mean_x fcout[(y*512+x)*512 + c]; writes f32 and f16 copies.
// ---------------------------------------------------------------------------
__global__ void feat_mean_kernel(const _Float16* __restrict__ fcout,
                                 float* __restrict__ feat32,
                                 _Float16* __restrict__ feat16)
{
    const int tid = blockIdx.x * blockDim.x + threadIdx.x;
    if (tid >= 32 * 512) return;
    const int yy = tid >> 9;
    const int c  = tid & 511;
    const _Float16* p = fcout + (size_t)yy * 512 * 512 + c;
    float s = 0.0f;
    for (int x = 0; x < 512; ++x) s += (float)p[(size_t)x * 512];
    s *= (1.0f / 512.0f);
    feat32[tid] = s;
    feat16[tid] = (_Float16)s;
}

// ---------------------------------------------------------------------------
// GRU scan (one block per direction, 384 threads). gx: [32][384] precomputed
// (input gates incl. bias). Accumulates the per-direction time-mean into
// hbar[dir*128 + c]; the reference mean over t makes ordering irrelevant.
// ---------------------------------------------------------------------------
__global__ void gru_scan_kernel(const float* __restrict__ gxf, const float* __restrict__ gxb,
                                const float* __restrict__ whhf, const float* __restrict__ bhhf,
                                const float* __restrict__ whhb, const float* __restrict__ bhhb,
                                float* __restrict__ hbar)
{
    __shared__ float h[128];
    __shared__ float gh[384];
    __shared__ float hsum[128];
    const int dir = blockIdx.x;
    const float* gx  = dir ? gxb  : gxf;
    const float* whh = dir ? whhb : whhf;
    const float* bhh = dir ? bhhb : bhhf;
    const int tid = threadIdx.x;
    if (tid < 128) { h[tid] = 0.0f; hsum[tid] = 0.0f; }
    __syncthreads();
    for (int step = 0; step < 32; ++step) {
        const int t = dir ? (31 - step) : step;  // backward scans feat[::-1]
        float s = bhh[tid];
        const float* wr = whh + tid * 128;
        for (int k = 0; k < 128; ++k) s += wr[k] * h[k];
        gh[tid] = s;
        __syncthreads();
        if (tid < 128) {
            const float* g = gx + t * 384;
            const float r  = 1.0f / (1.0f + __expf(-(g[tid] + gh[tid])));
            const float z  = 1.0f / (1.0f + __expf(-(g[128 + tid] + gh[128 + tid])));
            const float nn = tanhf(g[256 + tid] + r * gh[256 + tid]);
            const float hn = (1.0f - z) * nn + z * h[tid];
            h[tid] = hn;
            hsum[tid] += hn;
        }
        __syncthreads();
    }
    if (tid < 128) hbar[dir * 128 + tid] = hsum[tid] * (1.0f / 32.0f);
}

__global__ void classifier_kernel(const float* __restrict__ hbar,
                                  const float* __restrict__ clsW,
                                  const float* __restrict__ clsb,
                                  float* __restrict__ out)
{
    const int t = threadIdx.x;
    if (t < 5) {
        float s = clsb[t];
        for (int c = 0; c < 256; ++c) s += clsW[t * 256 + c] * hbar[c];
        out[t] = s;
    }
}

// ---------------------------------------------------------------------------
extern "C" void kernel_launch(void* const* d_in, const int* in_sizes, int n_in,
                              void* d_out, int out_size, void* d_ws, size_t ws_size,
                              hipStream_t stream)
{
    (void)in_sizes; (void)n_in; (void)out_size; (void)ws_size;
    const float* waveform = (const float*)d_in[0];
    const float* logmel   = (const float*)d_in[1];
    const float* wc1 = (const float*)d_in[2];   const float* wb1 = (const float*)d_in[3];
    const float* wc2 = (const float*)d_in[4];   const float* wb2 = (const float*)d_in[5];
    const float* wc3 = (const float*)d_in[6];   const float* wb3 = (const float*)d_in[7];
    const float* wc4 = (const float*)d_in[8];   const float* wb4 = (const float*)d_in[9];
    const float* mc1w = (const float*)d_in[10]; const float* mb1 = (const float*)d_in[11];
    const float* bn1g = (const float*)d_in[12]; const float* bn1b = (const float*)d_in[13];
    const float* bn1m = (const float*)d_in[14]; const float* bn1v = (const float*)d_in[15];
    const float* mc2w = (const float*)d_in[16]; const float* mb2 = (const float*)d_in[17];
    const float* bn2g = (const float*)d_in[18]; const float* bn2b = (const float*)d_in[19];
    const float* bn2m = (const float*)d_in[20]; const float* bn2v = (const float*)d_in[21];
    const float* mc3w = (const float*)d_in[22]; const float* mb3 = (const float*)d_in[23];
    const float* fcw = (const float*)d_in[24];  const float* fb  = (const float*)d_in[25];
    const float* wih_f = (const float*)d_in[26]; const float* whh_f = (const float*)d_in[27];
    const float* bih_f = (const float*)d_in[28]; const float* bhh_f = (const float*)d_in[29];
    const float* wih_b = (const float*)d_in[30]; const float* whh_b = (const float*)d_in[31];
    const float* bih_b = (const float*)d_in[32]; const float* bhh_b = (const float*)d_in[33];
    const float* clsW = (const float*)d_in[34]; const float* clsb = (const float*)d_in[35];
    float* out = (float*)d_out;

    char* ws = (char*)d_ws;
    size_t off = 0;
    auto alloc = [&](size_t bytes) -> char* {
        char* p = ws + off;
        off = (off + bytes + 255) & ~(size_t)255;
        return p;
    };

    // wave branch (f32 [C][L])
    float* p1 = (float*)alloc((size_t)64 * 800 * 4);
    float* p2 = (float*)alloc((size_t)128 * 200 * 4);
    float* p3 = (float*)alloc((size_t)256 * 50 * 4);
    float* c4 = (float*)alloc((size_t)256 * 50 * 4);
    // mel branch activations (NHWC f16)
    _Float16* act1 = (_Float16*)alloc((size_t)65536 * 64 * 2);
    _Float16* act2 = (_Float16*)alloc((size_t)65536 * 128 * 2);
    _Float16* act3 = (_Float16*)alloc((size_t)65536 * 256 * 2);
    _Float16* cat  = (_Float16*)alloc((size_t)16384 * 512 * 2);
    _Float16* fco  = (_Float16*)alloc((size_t)16384 * 512 * 2);
    // repacked f16 weights [tap][O][I]
    _Float16* w2t  = (_Float16*)alloc((size_t)9 * 128 * 64 * 2);
    _Float16* w3t  = (_Float16*)alloc((size_t)256 * 128 * 2);
    _Float16* wfct = (_Float16*)alloc((size_t)9 * 512 * 512 * 2);
    _Float16* wihf16 = (_Float16*)alloc((size_t)384 * 512 * 2);
    _Float16* wihb16 = (_Float16*)alloc((size_t)384 * 512 * 2);
    // epilogue tables, feat, gru
    float* ss = (float*)alloc((size_t)2944 * 4);
    float* feat32 = (float*)alloc((size_t)32 * 512 * 4);
    _Float16* feat16 = (_Float16*)alloc((size_t)32 * 512 * 2);
    float* gxf = (float*)alloc((size_t)32 * 384 * 4);
    float* gxb = (float*)alloc((size_t)32 * 384 * 4);
    float* hbar = (float*)alloc((size_t)256 * 4);

    // --- weight repack + epilogue tables ---
    convert_w_kernel<<<(128 * 64 * 9 + 255) / 256, 256, 0, stream>>>(mc2w, w2t, 128, 64, 3, 3);
    convert_w_kernel<<<(256 * 128 + 255) / 256, 256, 0, stream>>>(mc3w, w3t, 256, 128, 1, 1);
    convert_w_kernel<<<(512 * 512 * 9 + 255) / 256, 256, 0, stream>>>(fcw, wfct, 512, 512, 3, 3);
    convert_w_kernel<<<(384 * 512 + 255) / 256, 256, 0, stream>>>(wih_f, wihf16, 384, 512, 1, 1);
    convert_w_kernel<<<(384 * 512 + 255) / 256, 256, 0, stream>>>(wih_b, wihb16, 384, 512, 1, 1);
    setup_ss_kernel<<<1, 512, 0, stream>>>(mb2, bn2g, bn2b, bn2m, bn2v, mb3, fb, bih_f, bih_b, ss);

    // --- wave branch ---
    wave_conv_pool_kernel<<<(64 * 800 + 255) / 256, 256, 0, stream>>>(
        waveform, wc1, wb1, p1, 1, 16000, 64, 3200, 11, 5, 3, 4);
    wave_conv_pool_kernel<<<(128 * 200 + 255) / 256, 256, 0, stream>>>(
        p1, wc2, wb2, p2, 64, 800, 128, 800, 5, 1, 2, 4);
    wave_conv_pool_kernel<<<(256 * 50 + 255) / 256, 256, 0, stream>>>(
        p2, wc3, wb3, p3, 128, 200, 256, 200, 5, 1, 2, 4);
    wave_conv_pool_kernel<<<(256 * 50 + 255) / 256, 256, 0, stream>>>(
        p3, wc4, wb4, c4, 256, 50, 256, 50, 5, 1, 2, 1);

    // --- mel branch ---
    mel_conv1_kernel<<<(64 * 1024 * 64 + 255) / 256, 256, 0, stream>>>(
        logmel, mc1w, mb1, bn1g, bn1b, bn1m, bn1v, act1);
    // mc2: 3x3, 64->128, fused BN+ReLU  (M=65536, N=128, K=576)
    conv_wmma_kernel<<<dim3(512, 2), 128, 0, stream>>>(
        act1, w2t, ss + 0, ss + 128, act2, 64, 1024, 64, 128, 9, 1, 1);
    // mc3: 1x1, 128->256, bias+ReLU  (M=65536, N=256, K=128)
    conv_wmma_kernel<<<dim3(512, 4), 128, 0, stream>>>(
        act2, w3t, ss + 256, ss + 512, act3, 64, 1024, 128, 256, 1, 1, 1);

    // --- fuse: resize+pool+concat, then fc conv ---
    build_concat_kernel<<<(32 * 512 * 512 + 255) / 256, 256, 0, stream>>>(c4, act3, cat);
    // fc: 3x3, 512->512, bias+ReLU  (M=16384, N=512, K=4608) — the 77 GFLOP core
    conv_wmma_kernel<<<dim3(128, 8), 128, 0, stream>>>(
        cat, wfct, ss + 768, ss + 1280, fco, 32, 512, 512, 512, 9, 1, 1);

    // --- temporal features + GRU input gates (GEMM M=32, N=384, K=512) ---
    feat_mean_kernel<<<(32 * 512 + 255) / 256, 256, 0, stream>>>(fco, feat32, feat16);
    conv_wmma_kernel<<<dim3(1, 6), 128, 0, stream>>>(
        feat16, wihf16, ss + 1792, ss + 2176, gxf, 32, 1, 512, 384, 1, 0, 0);
    conv_wmma_kernel<<<dim3(1, 6), 128, 0, stream>>>(
        feat16, wihb16, ss + 1792, ss + 2560, gxb, 32, 1, 512, 384, 1, 0, 0);

    // --- GRU scan (both directions) + classifier ---
    gru_scan_kernel<<<2, 384, 0, stream>>>(gxf, gxb, whh_f, bhh_f, whh_b, bhh_b, hbar);
    classifier_kernel<<<1, 64, 0, stream>>>(hbar, clsW, clsb, out);
}
